// MultiHeadAttention_43611097924070
// MI455X (gfx1250) — matrix operand order repaired and checked by reading, not compiled, not run on hardware
//
#include <hip/hip_runtime.h>
#include <hip/hip_bf16.h>

// ---------------------------------------------------------------------------
// MHA forward for MI455X (gfx1250, wave32, WMMA).
//
// Math (after accounting for the reference's view-style reshapes):
//   Pq = q @ Wq^T + bq            [4096,1024]   (M = B*L)
//   Pk = k @ Wk^T + bk
//   Pv = v @ Wv^T + bv
//   per (b,h) flat chunk of 131072 floats:
//     Q = chunk[2048,64], KT = chunk[64,2048] (same memory!), V = chunk[2048,64]
//     S = (Q @ KT) / 8 ; P = softmax(S) ; O = P @ V   (flash-style, streamed)
//   out = O @ Wo^T + bo           [4096,1024]
//
// Precision: fp32-accurate via split bf16 (x = hi + lo), 3 bf16 WMMAs per
// K=32 chunk (AhBh + AlBh + AhBl) -> ~2.7x the f32-WMMA (16x16x4) rate.
// GEMM double-buffers LDS tiles so global traffic overlaps WMMA.
// ---------------------------------------------------------------------------

typedef __attribute__((ext_vector_type(16))) __bf16 v16bf;
typedef __attribute__((ext_vector_type(8)))  float  v8f;
typedef unsigned short u16;

static __device__ __forceinline__ u16 f2bf(float x) {
  unsigned u = __float_as_uint(x);
  return (u16)((u + 0x7FFFu + ((u >> 16) & 1u)) >> 16);  // round-to-nearest-even
}
static __device__ __forceinline__ float bf2f(u16 h) {
  return __uint_as_float((unsigned)h << 16);
}
static __device__ __forceinline__ void split_bf(float x, u16& h, u16& l) {
  h = f2bf(x);
  l = f2bf(x - bf2f(h));
}

// Build a v16bf fragment from two 16-byte LDS chunks (p and p+delta2 bytes).
static __device__ __forceinline__ v16bf frag_ld(const u16* p, int delta2) {
  float4 a = *reinterpret_cast<const float4*>(p);
  float4 b = *reinterpret_cast<const float4*>(reinterpret_cast<const char*>(p) + delta2);
  union { float4 f[2]; v16bf v; } u;
  u.f[0] = a; u.f[1] = b;
  return u.v;
}

static __device__ __forceinline__ v8f wmma_bf(v16bf a, v16bf b, v8f c) {
  return __builtin_amdgcn_wmma_f32_16x16x32_bf16(false, a, false, b, (short)0, c,
                                                 false, false);
}

// ---------------------------------------------------------------------------
// C[M,N] = A[M,K] @ W[N,K]^T + bias[N]   (both operands K-contiguous, "NT")
// Block 128x128, 8 waves; wave tile 32x64 = 2x4 WMMA tiles.
// Double-buffered LDS (80KB) + register prefetch: global loads of chunk i+1
// issue right after the single per-chunk barrier and overlap the WMMAs.
// ---------------------------------------------------------------------------
static constexpr int BM = 128, BN = 128, BK = 32, GPAD = 8, LDSK = BK + GPAD;

static __device__ __forceinline__ void load_tile(
    const float* __restrict__ A, const float* __restrict__ W,
    int m0, int n0, int K, int kb, int r0, int k4, float4* ra, float4* rb) {
#pragma unroll
  for (int it = 0; it < 4; ++it) {
    int row = it * 32 + r0;
    ra[it] = *reinterpret_cast<const float4*>(A + (size_t)(m0 + row) * K + kb + k4);
    rb[it] = *reinterpret_cast<const float4*>(W + (size_t)(n0 + row) * K + kb + k4);
  }
}

static __device__ __forceinline__ void store_tile(
    u16* sAh, u16* sAl, u16* sBh, u16* sBl,
    int r0, int k4, const float4* ra, const float4* rb) {
#pragma unroll
  for (int it = 0; it < 4; ++it) {
    int row = it * 32 + r0;
    float as[4] = {ra[it].x, ra[it].y, ra[it].z, ra[it].w};
    float bs[4] = {rb[it].x, rb[it].y, rb[it].z, rb[it].w};
    u16 ah[4], al[4], bh[4], bl[4];
#pragma unroll
    for (int j = 0; j < 4; ++j) { split_bf(as[j], ah[j], al[j]); split_bf(bs[j], bh[j], bl[j]); }
    uint2 p;
    p.x = ah[0] | ((unsigned)ah[1] << 16); p.y = ah[2] | ((unsigned)ah[3] << 16);
    *reinterpret_cast<uint2*>(&sAh[row * LDSK + k4]) = p;
    p.x = al[0] | ((unsigned)al[1] << 16); p.y = al[2] | ((unsigned)al[3] << 16);
    *reinterpret_cast<uint2*>(&sAl[row * LDSK + k4]) = p;
    p.x = bh[0] | ((unsigned)bh[1] << 16); p.y = bh[2] | ((unsigned)bh[3] << 16);
    *reinterpret_cast<uint2*>(&sBh[row * LDSK + k4]) = p;
    p.x = bl[0] | ((unsigned)bl[1] << 16); p.y = bl[2] | ((unsigned)bl[3] << 16);
    *reinterpret_cast<uint2*>(&sBl[row * LDSK + k4]) = p;
  }
}

__global__ __launch_bounds__(256) void gemm_nt_bias(
    const float* __restrict__ A, const float* __restrict__ W,
    const float* __restrict__ bias, float* __restrict__ C,
    int M, int N, int K) {
  __shared__ u16 sAh[2][BM * LDSK], sAl[2][BM * LDSK];
  __shared__ u16 sBh[2][BN * LDSK], sBl[2][BN * LDSK];

  const int tid  = threadIdx.x;
  const int lane = tid & 31;
  const int wave = tid >> 5;
  const int wm   = wave & 3;   // 4 waves along M
  const int wn   = wave >> 2;  // 2 waves along N
  const int lr   = lane & 15;
  const int sel  = lane >> 4;
  const int m0   = blockIdx.y * BM;
  const int n0   = blockIdx.x * BN;
  const int r0   = tid >> 3;          // staging row within 32-row group
  const int k4   = (tid & 7) * 4;     // staging k offset

  v8f acc[2][4] = {};
  float4 ra[4], rb[4];

  load_tile(A, W, m0, n0, K, 0, r0, k4, ra, rb);

  const int nch = K / BK;
  for (int i = 0; i < nch; ++i) {
    const int buf = i & 1;
    // buf was last read in chunk i-2; chunk i-1's barrier guarantees safety.
    store_tile(sAh[buf], sAl[buf], sBh[buf], sBl[buf], r0, k4, ra, rb);
    __syncthreads();
    if (i + 1 < nch)  // prefetch next tile into registers; overlaps WMMAs below
      load_tile(A, W, m0, n0, K, (i + 1) * BK, r0, k4, ra, rb);

    // ---- fragments (A: lane=row, K = sel*8..+7 then 16+sel*8..; B: lane=col, same K order)
    v16bf fah[2], fal[2], fbh[4], fbl[4];
#pragma unroll
    for (int mt = 0; mt < 2; ++mt) {
      int r = wm * 32 + mt * 16 + lr;
      fah[mt] = frag_ld(&sAh[buf][r * LDSK + sel * 8], 32);
      fal[mt] = frag_ld(&sAl[buf][r * LDSK + sel * 8], 32);
    }
#pragma unroll
    for (int nt = 0; nt < 4; ++nt) {
      int n = wn * 64 + nt * 16 + lr;
      fbh[nt] = frag_ld(&sBh[buf][n * LDSK + sel * 8], 32);
      fbl[nt] = frag_ld(&sBl[buf][n * LDSK + sel * 8], 32);
    }
    // ---- term-major 3-term split MACs: consecutive WMMAs hit distinct accums
#pragma unroll
    for (int mt = 0; mt < 2; ++mt)
#pragma unroll
      for (int nt = 0; nt < 4; ++nt) acc[mt][nt] = wmma_bf(fah[mt], fbh[nt], acc[mt][nt]);
#pragma unroll
    for (int mt = 0; mt < 2; ++mt)
#pragma unroll
      for (int nt = 0; nt < 4; ++nt) acc[mt][nt] = wmma_bf(fal[mt], fbh[nt], acc[mt][nt]);
#pragma unroll
    for (int mt = 0; mt < 2; ++mt)
#pragma unroll
      for (int nt = 0; nt < 4; ++nt) acc[mt][nt] = wmma_bf(fah[mt], fbl[nt], acc[mt][nt]);
  }

  // ---- epilogue: bias add, f32 store (C layout: row = v + 8*sel, col = lane&15)
#pragma unroll
  for (int mt = 0; mt < 2; ++mt) {
    int row = m0 + wm * 32 + mt * 16 + sel * 8;
#pragma unroll
    for (int nt = 0; nt < 4; ++nt) {
      int col = n0 + wn * 64 + nt * 16 + lr;
      float bv = bias[col];
      float* cp = C + (size_t)row * N + col;
#pragma unroll
      for (int v = 0; v < 8; ++v) cp[(size_t)v * N] = acc[mt][nt][v] + bv;
    }
  }
}

// ---------------------------------------------------------------------------
// Flash attention over one (b,h) flat chunk. 4 waves/block, 16 query rows per
// wave, full d=64 accumulator in registers; streams 64 key columns/iteration.
// Per-head K/V working set is ~1MB -> re-reads across the 32 row-blocks of a
// head are L2 (192MB) hits; global_prefetch_b8 pulls the next chunk early.
// ---------------------------------------------------------------------------
__global__ __launch_bounds__(128) void attention_flash(
    const float* __restrict__ Pq, const float* __restrict__ Pk,
    const float* __restrict__ Pv, float* __restrict__ O) {
  constexpr int L = 2048, DH = 64, NC = 64, PAD = 8;
  constexpr int KS = DH + PAD;   // 72 u16 = 144B rows (16B-aligned)
  constexpr int PS = NC + PAD;   // 72
  __shared__ u16 sKh[NC * KS], sKl[NC * KS];       // [key col n][kk]  (KT transposed)
  __shared__ u16 sVh[DH * KS], sVl[DH * KS];       // [d n][key kk]    (V transposed)
  __shared__ u16 sPh[4 * 16 * PS], sPl[4 * 16 * PS];

  const int tid  = threadIdx.x;
  const int lane = tid & 31;
  const int wave = tid >> 5;
  const int lr   = lane & 15;
  const int sel  = lane >> 4;
  const int head = blockIdx.x >> 5;  // b*16 + h, 0..31
  const int rb   = blockIdx.x & 31;  // query row block
  const size_t hoff = (size_t)head * (L * DH);
  const float* Qh = Pq + hoff;       // [2048,64]
  const float* Kt = Pk + hoff;       // same memory viewed [64,2048]
  const float* Vh = Pv + hoff;       // [2048,64]
  float* Oh = O + hoff;
  const int row0 = rb * 64 + wave * 16;

  // ---- load Q fragments once (scale 1/sqrt(64)=0.125 folded in, exact)
  v16bf qfh[2], qfl[2];
#pragma unroll
  for (int kc = 0; kc < 2; ++kc) {
    const float* qp = Qh + (size_t)(row0 + lr) * DH + kc * 32 + sel * 8;
    union { u16 s[16]; v16bf v; } uh, ul;
#pragma unroll
    for (int i = 0; i < 8; ++i) {
      float x0 = qp[i] * 0.125f;       split_bf(x0, uh.s[i],     ul.s[i]);
      float x1 = qp[16 + i] * 0.125f;  split_bf(x1, uh.s[8 + i], ul.s[8 + i]);
    }
    qfh[kc] = uh.v; qfl[kc] = ul.v;
  }

  v8f oacc[4] = {};
  float mrow[8], lrow[8];
#pragma unroll
  for (int i = 0; i < 8; ++i) { mrow[i] = -3.0e38f; lrow[i] = 0.f; }
  constexpr float L2E = 1.4426950408889634f;

  for (int c0 = 0; c0 < L; c0 += NC) {
    // ---- stage KT[:,c0:c0+64] and V[c0:c0+64,:] transposed as bf16 hi/lo
#pragma unroll
    for (int it = 0; it < 8; ++it) {
      int id = it * 128 + tid;   // 0..1023 float4s
      int r  = id >> 4;          // 0..63
      int c4 = (id & 15) * 4;    // 0..60
      float4 kx = *reinterpret_cast<const float4*>(Kt + (size_t)r * L + c0 + c4);
      float4 vx = *reinterpret_cast<const float4*>(Vh + (size_t)(c0 + r) * DH + c4);
      float ks4[4] = {kx.x, kx.y, kx.z, kx.w};
      float vs4[4] = {vx.x, vx.y, vx.z, vx.w};
#pragma unroll
      for (int j = 0; j < 4; ++j) {
        u16 h, l;
        split_bf(ks4[j], h, l);
        sKh[(c4 + j) * KS + r] = h;  sKl[(c4 + j) * KS + r] = l;
        split_bf(vs4[j], h, l);
        sVh[(c4 + j) * KS + r] = h;  sVl[(c4 + j) * KS + r] = l;
      }
    }
    // ---- prefetch next chunk's K/V lines (global_prefetch_b8)
    if (c0 + NC < L) {
      int r = tid & 63, part = tid >> 6;   // 128 threads: 2 cachelines per row set
      __builtin_prefetch(Kt + (size_t)r * L + (c0 + NC) + part * 32, 0, 1);
      __builtin_prefetch(Vh + (size_t)(c0 + NC + r) * DH + part * 32, 0, 1);
    }
    __syncthreads();

    // ---- S tiles: 16 x 64 per wave
    v8f st[4] = {};
#pragma unroll
    for (int kc = 0; kc < 2; ++kc) {
#pragma unroll
      for (int nt = 0; nt < 4; ++nt) {
        const u16* pbh = &sKh[(nt * 16 + lr) * KS + kc * 32 + sel * 8];
        const u16* pbl = &sKl[(nt * 16 + lr) * KS + kc * 32 + sel * 8];
        v16bf kb_h = frag_ld(pbh, 32);
        v16bf kb_l = frag_ld(pbl, 32);
        st[nt] = wmma_bf(qfh[kc], kb_h, st[nt]);
        st[nt] = wmma_bf(qfl[kc], kb_h, st[nt]);
        st[nt] = wmma_bf(qfh[kc], kb_l, st[nt]);
      }
    }

    // ---- online softmax (row = v + 8*sel; reduce across 16 lanes of the half)
    float fscale[8];
#pragma unroll
    for (int v = 0; v < 8; ++v) {
      float x = fmaxf(fmaxf(st[0][v], st[1][v]), fmaxf(st[2][v], st[3][v]));
      x = fmaxf(x, __shfl_xor(x, 1, 32));
      x = fmaxf(x, __shfl_xor(x, 2, 32));
      x = fmaxf(x, __shfl_xor(x, 4, 32));
      x = fmaxf(x, __shfl_xor(x, 8, 32));
      float mnew = fmaxf(mrow[v], x);
      fscale[v] = exp2f((mrow[v] - mnew) * L2E);
      mrow[v] = mnew;
    }
#pragma unroll
    for (int v = 0; v < 8; ++v) {
      float s = 0.f;
#pragma unroll
      for (int nt = 0; nt < 4; ++nt) {
        float p = exp2f((st[nt][v] - mrow[v]) * L2E);
        st[nt][v] = p;
        s += p;
      }
      s += __shfl_xor(s, 1, 32);
      s += __shfl_xor(s, 2, 32);
      s += __shfl_xor(s, 4, 32);
      s += __shfl_xor(s, 8, 32);
      lrow[v] = lrow[v] * fscale[v] + s;
    }
#pragma unroll
    for (int t = 0; t < 4; ++t)
#pragma unroll
      for (int v = 0; v < 8; ++v) oacc[t][v] *= fscale[v];

    // ---- bounce P through LDS (C layout -> A-fragment layout), bf16 hi/lo.
    // P region is per-wave private: same-wave LDS ops are in-order, no barrier.
    u16* ph = sPh + wave * (16 * PS);
    u16* pl = sPl + wave * (16 * PS);
#pragma unroll
    for (int nt = 0; nt < 4; ++nt)
#pragma unroll
      for (int v = 0; v < 8; ++v) {
        int r = v + sel * 8, c = nt * 16 + lr;
        u16 h, l;
        split_bf(st[nt][v], h, l);
        ph[r * PS + c] = h;
        pl[r * PS + c] = l;
      }

    // ---- O += P @ V
#pragma unroll
    for (int kc = 0; kc < 2; ++kc) {
      v16bf pah = frag_ld(ph + lr * PS + kc * 32 + sel * 8, 32);
      v16bf pal = frag_ld(pl + lr * PS + kc * 32 + sel * 8, 32);
#pragma unroll
      for (int t = 0; t < 4; ++t) {
        const u16* vbh = &sVh[(t * 16 + lr) * KS + kc * 32 + sel * 8];
        const u16* vbl = &sVl[(t * 16 + lr) * KS + kc * 32 + sel * 8];
        v16bf vh_ = frag_ld(vbh, 32);
        v16bf vl_ = frag_ld(vbl, 32);
        oacc[t] = wmma_bf(pah, vh_, oacc[t]);
        oacc[t] = wmma_bf(pal, vh_, oacc[t]);
        oacc[t] = wmma_bf(pah, vl_, oacc[t]);
      }
    }
    __syncthreads();   // all waves done with sK/sV before next staging
  }

  // ---- epilogue: O /= l, store f32 into flat head chunk
#pragma unroll
  for (int t = 0; t < 4; ++t) {
    int col = t * 16 + lr;
    float* op = Oh + (size_t)(row0 + sel * 8) * DH + col;
#pragma unroll
    for (int v = 0; v < 8; ++v) op[(size_t)v * DH] = oacc[t][v] / lrow[v];
  }
}

// ---------------------------------------------------------------------------
extern "C" void kernel_launch(void* const* d_in, const int* in_sizes, int n_in,
                              void* d_out, int out_size, void* d_ws, size_t ws_size,
                              hipStream_t stream) {
  const float* q  = (const float*)d_in[0];
  const float* k  = (const float*)d_in[1];
  const float* v  = (const float*)d_in[2];
  const float* wq = (const float*)d_in[3];
  const float* bq = (const float*)d_in[4];
  const float* wk = (const float*)d_in[5];
  const float* bk = (const float*)d_in[6];
  const float* wv = (const float*)d_in[7];
  const float* bv = (const float*)d_in[8];
  const float* wo = (const float*)d_in[9];
  const float* bo = (const float*)d_in[10];

  constexpr int M = 4096;   // B*L
  constexpr int D = 1024;
  float* Pq = (float*)d_ws;                 // 16 MB each
  float* Pk = Pq + (size_t)M * D;
  float* Pv = Pk + (size_t)M * D;
  float* Ob = Pv + (size_t)M * D;           // total ws use: 64 MB

  dim3 gg(D / 128, M / 128);                // 8 x 32 blocks
  gemm_nt_bias<<<gg, 256, 0, stream>>>(q, wq, bq, Pq, M, D, D);
  gemm_nt_bias<<<gg, 256, 0, stream>>>(k, wk, bk, Pk, M, D, D);
  gemm_nt_bias<<<gg, 256, 0, stream>>>(v, wv, bv, Pv, M, D, D);

  attention_flash<<<dim3(32 * 32), 128, 0, stream>>>(Pq, Pk, Pv, Ob);

  gemm_nt_bias<<<gg, 256, 0, stream>>>(Ob, wo, bo, (float*)d_out, M, D, D);
}
